// MSARowAttentionWithPairBias_16054587752601
// MI455X (gfx1250) — compile-verified, matrix operand order checked by reading, not run on hardware
//
#include <hip/hip_runtime.h>
#include <hip/hip_bf16.h>
#include <math.h>

// MSARowAttentionWithPairBias for MI455X (gfx1250, wave32, WMMA 16x16x32 f16)
// B=1, S=128, R=256, D_NODE=256, D_PAIR=128, H=8, C=32

typedef _Float16 f16;
typedef __attribute__((ext_vector_type(16))) _Float16 v16h;
typedef __attribute__((ext_vector_type(8)))  _Float16 v8h;
typedef __attribute__((ext_vector_type(8)))  float    v8f;

typedef unsigned int u32x4 __attribute__((ext_vector_type(4)));
typedef int          i32x4 __attribute__((ext_vector_type(4)));
typedef int          i32x8 __attribute__((ext_vector_type(8)));

#ifndef __has_builtin
#define __has_builtin(x) 0
#endif
#if __has_builtin(__builtin_amdgcn_tensor_load_to_lds) && \
    __has_builtin(__builtin_amdgcn_s_wait_tensorcnt)
#define USE_TDM 1
#else
#define USE_TDM 0
#endif

// Per-lane K-element order inside a 16-bit WMMA A/B fragment (ISA 7.12.2):
// lanes 0-15 hold K = {0..7, 16..23}, lanes 16-31 hold K = {8..15, 24..31}.
__device__ __forceinline__ int kk_of(int e, int hf) {
  int v = e >> 1, lo = e & 1;
  return ((v >> 2) << 4) + hf * 8 + ((v & 3) << 1) + lo;
}

__device__ __forceinline__ v8f wmma16(v16h a, v16h b, v8f c) {
  return __builtin_amdgcn_wmma_f32_16x16x32_f16(false, a, false, b, (short)0, c,
                                                false, false);
}

// Row-major f16 source: lane's 16 fragment halfs are two contiguous 8-half
// chunks at kbase + hf*8 and kbase + 16 + hf*8 of one row -> two b128 loads.
__device__ __forceinline__ v16h frag_rowmajor(const f16* base, int strideH,
                                              int row, int kbase, int hf) {
  const f16* p = base + (size_t)row * strideH + kbase + hf * 8;
  v8h lo = *(const v8h*)p;
  v8h hi = *(const v8h*)(p + 16);
  v16h r;
#pragma unroll
  for (int e = 0; e < 8; ++e) { r[e] = lo[e]; r[e + 8] = hi[e]; }
  return r;
}

// Column gather (for B = row-major V): fixed column, strided K rows.
__device__ __forceinline__ v16h frag_gather_col(const f16* base, int strideH,
                                                int col, int krow, int hf) {
  v16h r;
#pragma unroll
  for (int e = 0; e < 16; ++e)
    r[e] = base[(size_t)(krow + kk_of(e, hf)) * strideH + col];
  return r;
}

// ---------------------------------------------------------------------------
// Kernel 0: pack f32 weights [256 x 256] into f16 WMMA B-fragment order.
// Fragment buffer: [(kt*16+nt)*32 + lane]*16 halfs, kt in 0..7, nt in 0..15.
// ---------------------------------------------------------------------------
__global__ void __launch_bounds__(256) k_swizzle(
    const float* w0, const float* w1, const float* w2, const float* w3,
    const float* w4, f16* f0, f16* f1, f16* f2, f16* f3, f16* f4) {
  int t = blockIdx.x * blockDim.x + threadIdx.x;
  if (t >= 5 * 8 * 16 * 32) return;
  int w = t / 4096, rem = t % 4096;
  int kt = rem / 512; rem %= 512;
  int nt = rem / 32; int lane = rem % 32;
  const float* src = (w == 0) ? w0 : (w == 1) ? w1 : (w == 2) ? w2
                     : (w == 3) ? w3 : w4;
  f16* dst = ((w == 0) ? f0 : (w == 1) ? f1 : (w == 2) ? f2
              : (w == 3) ? f3 : f4) +
             ((size_t)(kt * 16 + nt) * 32 + lane) * 16;
  int hf = lane >> 4, col = nt * 16 + (lane & 15);
#pragma unroll
  for (int e = 0; e < 16; ++e)
    dst[e] = (f16)src[(kt * 32 + kk_of(e, hf)) * 256 + col];
}

// ---------------------------------------------------------------------------
// Kernel A: LayerNorm(M_raw) then fused Q/K/V/G projection GEMM via WMMA.
// One block = 16 rows of the [32768 x 256] activation, 8 waves x 8 tiles.
// q/k/v stored f16 head-major [s][h][r][32]; g stored f16 [row][256].
// ---------------------------------------------------------------------------
__global__ void __launch_bounds__(256) k_ln_qkvg(
    const float* M_raw, const float* g_ln, const float* b_ln, const float* bg,
    const f16* fq, const f16* fk, const f16* fv, const f16* fg,
    f16* q16, f16* k16, f16* v16p, f16* g16) {
  __shared__ float lraw[16][256];
  __shared__ float lstat[16][2];
  __shared__ f16   lx[16][256];
  int tid = threadIdx.x;
  size_t base = (size_t)blockIdx.x * 16 * 256;
#pragma unroll
  for (int it = 0; it < 16; ++it) {
    int idx = it * 256 + tid;
    lraw[idx >> 8][idx & 255] = M_raw[base + idx];
  }
  __syncthreads();
  if (tid < 16) {
    float s = 0.f;
    for (int c = 0; c < 256; ++c) s += lraw[tid][c];
    float mu = s * (1.f / 256.f);
    float vv = 0.f;
    for (int c = 0; c < 256; ++c) { float d = lraw[tid][c] - mu; vv += d * d; }
    lstat[tid][0] = mu;
    lstat[tid][1] = rsqrtf(vv * (1.f / 256.f) + 1e-5f);
  }
  __syncthreads();
#pragma unroll
  for (int it = 0; it < 16; ++it) {
    int idx = it * 256 + tid;
    int r = idx >> 8, c = idx & 255;
    lx[r][c] = (f16)((lraw[r][c] - lstat[r][0]) * lstat[r][1] * g_ln[c] + b_ln[c]);
  }
  __syncthreads();

  int wave = tid >> 5, lane = tid & 31;
  int hf = lane >> 4, colL = lane & 15;
  int mat = wave >> 1, ntb = (wave & 1) * 8;  // each wave: one matrix, 8 n-tiles
  const f16* fr = (mat == 0) ? fq : (mat == 1) ? fk : (mat == 2) ? fv : fg;
  int gr0 = blockIdx.x * 16;
#pragma unroll
  for (int t = 0; t < 8; ++t) {
    int nt = ntb + t;
    __builtin_prefetch(fr + ((size_t)(nt + 16) * 32 + lane) * 16, 0, 1);
    v8f acc = {0, 0, 0, 0, 0, 0, 0, 0};
#pragma unroll
    for (int kt = 0; kt < 8; ++kt) {
      v16h a = frag_rowmajor(&lx[0][0], 256, colL, kt * 32, hf);
      v16h b = *(const v16h*)(fr + ((size_t)(kt * 16 + nt) * 32 + lane) * 16);
      acc = wmma16(a, b, acc);
    }
    int n = nt * 16 + colL;
#pragma unroll
    for (int r = 0; r < 8; ++r) {
      int gr = gr0 + r + 8 * hf;   // C-layout: lanes 16-31 hold rows +8
      float val = acc[r];
      if (mat == 3) {
        val = 1.f / (1.f + __expf(-(val + bg[n])));
        g16[(size_t)gr * 256 + n] = (f16)val;
      } else {
        if (mat == 0) val *= 0.17677669529663687f;  // 1/sqrt(C)
        int s_ = gr >> 8, i_ = gr & 255;
        int h_ = n >> 5, c_ = n & 31;
        f16* dst = (mat == 0) ? q16 : (mat == 1) ? k16 : v16p;
        dst[(((size_t)s_ * 8 + h_) * 256 + i_) * 32 + c_] = (f16)val;
      }
    }
  }
}

// ---------------------------------------------------------------------------
// Kernel B: LayerNorm(Z) + pair bias pb[h][i][j] = Zn[i,j,:] . b_weights[h,:]
// One wave per (i,j) pair; wave-wide shuffle reductions over 128 channels.
// ---------------------------------------------------------------------------
__global__ void __launch_bounds__(256) k_pairbias(
    const float* Z, const float* gz, const float* bz, const float* bw,
    float* pb) {
  int wave = threadIdx.x >> 5, lane = threadIdx.x & 31;
  int pair = blockIdx.x * 8 + wave;  // = i*256 + j
  const float* zp = Z + (size_t)pair * 128;
  float zv[4];
#pragma unroll
  for (int t = 0; t < 4; ++t) zv[t] = zp[lane + 32 * t];
  float s = zv[0] + zv[1] + zv[2] + zv[3];
#pragma unroll
  for (int d = 1; d < 32; d <<= 1) s += __shfl_xor(s, d, 32);
  float mu = s * (1.f / 128.f);
  float vv = 0.f;
#pragma unroll
  for (int t = 0; t < 4; ++t) { float dd = zv[t] - mu; vv += dd * dd; }
#pragma unroll
  for (int d = 1; d < 32; d <<= 1) vv += __shfl_xor(vv, d, 32);
  float rstd = rsqrtf(vv * (1.f / 128.f) + 1e-5f);
  float nv[4];
#pragma unroll
  for (int t = 0; t < 4; ++t) {
    int p = lane + 32 * t;
    nv[t] = (zv[t] - mu) * rstd * gz[p] + bz[p];
  }
#pragma unroll
  for (int h = 0; h < 8; ++h) {
    float dsum = 0.f;
#pragma unroll
    for (int t = 0; t < 4; ++t) dsum += nv[t] * bw[h * 128 + lane + 32 * t];
#pragma unroll
    for (int d = 1; d < 32; d <<= 1) dsum += __shfl_xor(dsum, d, 32);
    if (lane == 0) pb[(size_t)h * 65536 + pair] = dsum;
  }
}

// ---------------------------------------------------------------------------
// Kernel C: attention for one (s,h): logits = QK^T + pb + mask, softmax,
// O = attn @ V. C=32 -> one WMMA per 16x16 logits tile; 4 waves x 4 i-tiles.
// V tile (16 KB, [256][32] f16) staged into LDS by the Tensor Data Mover.
// ---------------------------------------------------------------------------
__global__ void __launch_bounds__(128) k_attention(
    const f16* q16, const f16* k16, const f16* v16p, const float* pb,
    const float* mask, float* o32) {
  __shared__ f16 lv[256 * 32];        // V tile, 16 KB
  __shared__ f16 lattn[4][16 * 256];  // per-wave attn staging, 32 KB
  int s_ = blockIdx.x >> 3, h_ = blockIdx.x & 7;
  size_t headoff = ((size_t)s_ * 8 + h_) * 256 * 32;
  const f16* qs = q16 + headoff;
  const f16* ks = k16 + headoff;
  const f16* vs = v16p + headoff;

#if USE_TDM
  if (threadIdx.x < 32) {
    // D# group 0 (ISA 8.3): count=1, lds_addr, 57-bit global addr, type=2.
    // Generic LDS pointers carry the LDS byte offset in addr[31:0].
    unsigned lds_off = (unsigned)(unsigned long long)(const void*)&lv[0];
    unsigned long long ga = (unsigned long long)(const void*)vs;
    u32x4 g0;
    g0[0] = 1u;                                   // count=1, user mode
    g0[1] = lds_off;                              // lds_addr
    g0[2] = (unsigned)(ga & 0xffffffffu);         // global_addr[31:0]
    g0[3] = (unsigned)((ga >> 32) & 0x01ffffffu)  // global_addr[56:32]
            | (2u << 30);                         // type=2 (image)
    // D# group 1 (ISA 8.4): data_size=2B, 2D tile 32 x 256, stride 32.
    i32x8 g1;
    g1[0] = (int)(1u << 16);    // workgroup_mask=0, data_size=1 (2 bytes)
    g1[1] = (int)(32u << 16);   // tensor_dim0[15:0]=32
    g1[2] = (int)(256u << 16);  // tensor_dim0 hi=0, tensor_dim1[15:0]=256
    g1[3] = (int)(32u << 16);   // tensor_dim1 hi=0, tile_dim0=32
    g1[4] = 256;                // tile_dim1=256, tile_dim2=0
    g1[5] = 32;                 // tensor_dim0_stride[31:0]=32
    g1[6] = 0;                  // stride hi, tensor_dim1_stride lo
    g1[7] = 0;
    i32x4 gz4 = {0, 0, 0, 0};   // groups 2/3 unused (<=2D tensor)
#if __clang_major__ >= 23
    i32x8 gz8 = {0, 0, 0, 0, 0, 0, 0, 0};
    __builtin_amdgcn_tensor_load_to_lds(g0, g1, gz4, gz4, gz8, 0);
#else
    __builtin_amdgcn_tensor_load_to_lds(g0, g1, gz4, gz4, 0);
#endif
    __builtin_amdgcn_s_wait_tensorcnt(0);
  }
#else
  for (int t = threadIdx.x; t < 8192; t += 128) lv[t] = vs[t];
#endif
  __syncthreads();

  int wave = threadIdx.x >> 5, lane = threadIdx.x & 31;
  int hf = lane >> 4, colL = lane & 15;
  const float* pbh = pb + (size_t)h_ * 65536;
  const float* mrow = mask + (size_t)s_ * 256;
  f16* la = &lattn[wave][0];

  for (int itb = 0; itb < 4; ++itb) {
    int i0 = (itb * 4 + wave) * 16;
    v16h qa = frag_rowmajor(qs, 32, i0 + colL, 0, hf);
    v8f acc[16];
#pragma unroll
    for (int jt = 0; jt < 16; ++jt) {
      v16h kb = frag_rowmajor(ks, 32, jt * 16 + colL, 0, hf);
      v8f z = {0, 0, 0, 0, 0, 0, 0, 0};
      acc[jt] = wmma16(qa, kb, z);
    }
    // add pair bias + mask bias
#pragma unroll
    for (int jt = 0; jt < 16; ++jt) {
      int j = jt * 16 + colL;
      float mb = 1e9f * (mrow[j] - 1.f);
#pragma unroll
      for (int r = 0; r < 8; ++r)
        acc[jt][r] += pbh[(size_t)(i0 + r + 8 * hf) * 256 + j] + mb;
    }
    // softmax over j: xor shuffles 1..8 stay within each 16-lane half,
    // matching the C-layout row split (lanes 0-15: M=r, 16-31: M=r+8).
    float rinv[8];
#pragma unroll
    for (int r = 0; r < 8; ++r) {
      float m = acc[0][r];
#pragma unroll
      for (int jt = 1; jt < 16; ++jt) m = fmaxf(m, acc[jt][r]);
#pragma unroll
      for (int d = 1; d < 16; d <<= 1) m = fmaxf(m, __shfl_xor(m, d, 32));
      float ssum = 0.f;
#pragma unroll
      for (int jt = 0; jt < 16; ++jt) {
        float e = __expf(acc[jt][r] - m);
        acc[jt][r] = e;
        ssum += e;
      }
#pragma unroll
      for (int d = 1; d < 16; d <<= 1) ssum += __shfl_xor(ssum, d, 32);
      rinv[r] = 1.f / ssum;
    }
    // stage unnormalized exp to LDS in row-major for A-fragment reload
#pragma unroll
    for (int jt = 0; jt < 16; ++jt)
#pragma unroll
      for (int r = 0; r < 8; ++r)
        la[(r + 8 * hf) * 256 + jt * 16 + colL] = (f16)acc[jt][r];
    // O = attn @ V  (1/rowsum folded into the epilogue)
#pragma unroll
    for (int nt = 0; nt < 2; ++nt) {
      v8f oacc = {0, 0, 0, 0, 0, 0, 0, 0};
#pragma unroll
      for (int kt = 0; kt < 8; ++kt) {
        v16h aa = frag_rowmajor(la, 256, colL, kt * 32, hf);
        v16h vb = frag_gather_col(lv, 32, nt * 16 + colL, kt * 32, hf);
        oacc = wmma16(aa, vb, oacc);
      }
#pragma unroll
      for (int r = 0; r < 8; ++r) {
        int i = i0 + r + 8 * hf;
        int n = h_ * 32 + nt * 16 + colL;
        o32[((size_t)s_ * 256 + i) * 256 + n] = oacc[r] * rinv[r];
      }
    }
  }
}

// ---------------------------------------------------------------------------
// Kernel D: out = (g * o) @ Wo + out_bias + M_raw
// ---------------------------------------------------------------------------
__global__ void __launch_bounds__(256) k_out(
    const f16* g16, const float* o32, const f16* fo, const float* out_bias,
    const float* M_raw, float* out) {
  __shared__ f16 lx[16][256];
  int tid = threadIdx.x;
  size_t base = (size_t)blockIdx.x * 16 * 256;
#pragma unroll
  for (int it = 0; it < 16; ++it) {
    int idx = it * 256 + tid;
    float gv = (float)g16[base + idx];
    lx[idx >> 8][idx & 255] = (f16)(gv * o32[base + idx]);
  }
  __syncthreads();
  int wave = tid >> 5, lane = tid & 31, hf = lane >> 4, colL = lane & 15;
  int gr0 = blockIdx.x * 16;
#pragma unroll
  for (int t = 0; t < 2; ++t) {
    int nt = wave * 2 + t;
    v8f acc = {0, 0, 0, 0, 0, 0, 0, 0};
#pragma unroll
    for (int kt = 0; kt < 8; ++kt) {
      v16h a = frag_rowmajor(&lx[0][0], 256, colL, kt * 32, hf);
      v16h b = *(const v16h*)(fo + ((size_t)(kt * 16 + nt) * 32 + lane) * 16);
      acc = wmma16(a, b, acc);
    }
    int n = nt * 16 + colL;
#pragma unroll
    for (int r = 0; r < 8; ++r) {
      size_t gi = (size_t)(gr0 + r + 8 * hf) * 256 + n;
      out[gi] = acc[r] + out_bias[n] + M_raw[gi];
    }
  }
}

// ---------------------------------------------------------------------------
extern "C" void kernel_launch(void* const* d_in, const int* in_sizes, int n_in,
                              void* d_out, int out_size, void* d_ws,
                              size_t ws_size, hipStream_t stream) {
  (void)in_sizes; (void)n_in; (void)out_size; (void)ws_size;
  const float* M_raw  = (const float*)d_in[0];
  const float* Z      = (const float*)d_in[1];
  const float* M_mask = (const float*)d_in[2];
  const float* ln_m_g = (const float*)d_in[3];
  const float* ln_m_b = (const float*)d_in[4];
  const float* ln_z_g = (const float*)d_in[5];
  const float* ln_z_b = (const float*)d_in[6];
  const float* b_w    = (const float*)d_in[7];
  const float* Wq     = (const float*)d_in[8];
  const float* Wk     = (const float*)d_in[9];
  const float* Wv     = (const float*)d_in[10];
  const float* Wg     = (const float*)d_in[11];
  const float* bg     = (const float*)d_in[12];
  const float* Wo     = (const float*)d_in[13];
  const float* ob     = (const float*)d_in[14];
  float* out = (float*)d_out;

  char* ws = (char*)d_ws;
  size_t off = 0;
  auto take = [&](size_t b) {
    char* p = ws + off;
    off += (b + 255) & ~(size_t)255;
    return p;
  };
  f16* fq   = (f16*)take(131072);      // swizzled weight fragments (f16)
  f16* fk   = (f16*)take(131072);
  f16* fv   = (f16*)take(131072);
  f16* fg   = (f16*)take(131072);
  f16* fo   = (f16*)take(131072);
  f16* q16  = (f16*)take(16777216);    // [s][h][256][32] f16
  f16* k16  = (f16*)take(16777216);
  f16* v16p = (f16*)take(16777216);
  f16* g16  = (f16*)take(16777216);    // [row][256] f16
  float* o32 = (float*)take(33554432); // [row][256] f32
  float* pb  = (float*)take(2097152);  // [h][i][j] f32

  k_swizzle<<<80, 256, 0, stream>>>(Wq, Wk, Wv, Wg, Wo, fq, fk, fv, fg, fo);
  k_ln_qkvg<<<2048, 256, 0, stream>>>(M_raw, ln_m_g, ln_m_b, bg, fq, fk, fv,
                                      fg, q16, k16, v16p, g16);
  k_pairbias<<<8192, 256, 0, stream>>>(Z, ln_z_g, ln_z_b, b_w, pb);
  k_attention<<<1024, 128, 0, stream>>>(q16, k16, v16p, pb, M_mask, o32);
  k_out<<<2048, 256, 0, stream>>>(g16, o32, fo, ob, M_raw, out);
}
